// EvaAttention_29738353557841
// MI455X (gfx1250) — compile-verified, matrix operand order checked by reading, not run on hardware
//
#include <hip/hip_runtime.h>

// EVA attention for MI455X (gfx1250, wave32, WMMA).
// f32->bf16 convert; QKV GEMM (32x32 register-blocked WMMA) fused with bias,
// interleaved-pair RoPE (lane-pair shfl_xor) and 1/sqrt(D) scale; flash-style
// SDPA with async global->LDS K staging (ASYNCcnt), online softmax, and
// V kept transposed in memory so P.V B-fragments are contiguous b128 loads;
// output projection GEMM (32x32 register-blocked WMMA).

typedef __attribute__((ext_vector_type(16))) __bf16 bf16x16;
typedef __attribute__((ext_vector_type(8)))  __bf16 bf16x8;
typedef __attribute__((ext_vector_type(8)))  float  f32x8;

#define NUM_H   12
#define HD      64
#define NTOK    577
#define BATCH   32
#define CH      768
#define M_ROWS  (BATCH * NTOK)   // 18464, divisible by 32
#define NPAD    608              // 19 * 32 : padded token count for K-loop
#define QTILES  37               // ceil(577/16)

// 16-bit A/B fragment (row-major, K contiguous), ISA 7.12.2 layout:
// lane group g = lane>>4 ; VGPR0..3 hold K = 8g..8g+7, VGPR4..7 hold K = 16+8g..16+8g+7.
__device__ __forceinline__ bf16x16 load_frag16(const __bf16* p) {
    bf16x16 f;
    *(bf16x8*)&f       = *(const bf16x8*)(p);        // K = kb .. kb+7
    *((bf16x8*)&f + 1) = *(const bf16x8*)(p + 16);   // K = kb+16 .. kb+23
    return f;
}

__device__ __forceinline__ f32x8 wmma_bf16(bf16x16 a, bf16x16 b, f32x8 c) {
    return __builtin_amdgcn_wmma_f32_16x16x32_bf16(false, a, false, b,
                                                   (short)0, c, false, false);
}

// 32x32 output per wave: acc[mt*2+ot] = A[m0+mt*16, :] . W[o0+ot*16, :]^T
// 4 WMMA per 8 b128 loads (2x the compute density of a single-tile wave).
__device__ __forceinline__ void gemm_tile2x2(const __bf16* __restrict__ A,
                                             const __bf16* __restrict__ W,
                                             int m0, int o0, int K, int lane,
                                             f32x8 acc[4]) {
    const int g8 = (lane >> 4) << 3;
    const __bf16* ar0 = A + (long)(m0 + (lane & 15)) * K + g8;
    const __bf16* ar1 = ar0 + (long)16 * K;
    const __bf16* wr0 = W + (long)(o0 + (lane & 15)) * K + g8;
    const __bf16* wr1 = wr0 + (long)16 * K;
    for (int kc = 0; kc < K; kc += 32) {
        bf16x16 a0 = load_frag16(ar0 + kc);
        bf16x16 a1 = load_frag16(ar1 + kc);
        bf16x16 b0 = load_frag16(wr0 + kc);
        bf16x16 b1 = load_frag16(wr1 + kc);
        acc[0] = wmma_bf16(a0, b0, acc[0]);
        acc[1] = wmma_bf16(a0, b1, acc[1]);
        acc[2] = wmma_bf16(a1, b0, acc[2]);
        acc[3] = wmma_bf16(a1, b1, acc[3]);
    }
}

// ---------------- conversion / init ----------------
__global__ void cvt_f32_bf16(const float* __restrict__ in, __bf16* __restrict__ out, long n) {
    long i = (long)blockIdx.x * blockDim.x + threadIdx.x;
    long stride = (long)gridDim.x * blockDim.x;
    for (; i < n; i += stride) out[i] = (__bf16)in[i];
}

__global__ void zero_bf16(__bf16* __restrict__ p, long n) {
    long i = (long)blockIdx.x * blockDim.x + threadIdx.x;
    long stride = (long)gridDim.x * blockDim.x;
    for (; i < n; i += stride) p[i] = (__bf16)0.f;
}

// ---------------- QKV GEMM + bias + RoPE + scale + scatter ----------------
// grid: (M_ROWS/32, 3*CH/256), block 256 (8 waves); wave -> one 32x32 tile.
__global__ void qkv_rope_kernel(const __bf16* __restrict__ xB,
                                const __bf16* __restrict__ wB,
                                const float* __restrict__ q_bias,
                                const float* __restrict__ v_bias,
                                const float* __restrict__ rope,   // [576][128] = cat(sin,cos)
                                __bf16* __restrict__ Qb,          // [B,H,NPAD,HD]
                                __bf16* __restrict__ Kb,          // [B,H,NPAD,HD]
                                __bf16* __restrict__ VbT) {       // [B,H,HD,NPAD] (transposed)
    const int lane = threadIdx.x & 31;
    const int wave = threadIdx.x >> 5;
    const int m0 = blockIdx.x * 32;
    const int o0 = (blockIdx.y * 8 + wave) * 32;

    f32x8 acc[4] = {};
    gemm_tile2x2(xB, wB, m0, o0, CH, lane, acc);

    const int rowoff = (lane >> 4) << 3;
#pragma unroll
    for (int ot = 0; ot < 2; ++ot) {
        const int col = o0 + ot * 16 + (lane & 15);
        const int s  = col / CH;          // 0=q 1=k 2=v
        const int oo = col % CH;
        const int h  = oo >> 6;
        const int d  = oo & 63;
        const float bias = (s == 0) ? q_bias[oo] : ((s == 2) ? v_bias[oo] : 0.f);
#pragma unroll
        for (int mt = 0; mt < 2; ++mt) {
            f32x8 a = acc[mt * 2 + ot];
#pragma unroll
            for (int r = 0; r < 8; ++r) {
                const int m = m0 + mt * 16 + r + rowoff;
                const int b = m / NTOK;
                const int t = m - b * NTOK;
                float v = a[r] + bias;
                // pair partner (d ^ 1) lives in the adjacent lane of the C fragment
                float partner = __shfl_xor(v, 1, 32);
                if (s < 2 && t > 0) {
                    float rot  = (d & 1) ? partner : -partner;   // rot(x): (-x_odd, x_even)
                    float sine = rope[(t - 1) * (2 * HD) + d];
                    float cose = rope[(t - 1) * (2 * HD) + HD + d];
                    v = v * cose + rot * sine;
                }
                if (s == 0) v *= 0.125f;  // fold 1/sqrt(64) into Q
                const __bf16 bv = (__bf16)v;
                const int bh = b * NUM_H + h;
                if (s == 0)
                    Qb[(((long)bh) * NPAD + t) * HD + d] = bv;
                else if (s == 1)
                    Kb[(((long)bh) * NPAD + t) * HD + d] = bv;
                else
                    VbT[((long)bh * HD + d) * NPAD + t] = bv;   // transposed store
            }
        }
    }
}

// ---------------- flash attention: one wave per (b,h,q-tile) ----------------
__global__ void attn_kernel(const __bf16* __restrict__ Qb,
                            const __bf16* __restrict__ Kb,
                            const __bf16* __restrict__ VbT,
                            __bf16* __restrict__ Ob) {
    __shared__ __align__(16) __bf16 sP[8][16 * 32];        // per-wave P tile (16x32)
    __shared__ __align__(16) __bf16 sK[8][2][32 * HD];     // per-wave K tile, double-buffered

    const int lane = threadIdx.x & 31;
    const int wave = threadIdx.x >> 5;
    const int task = blockIdx.x * 8 + wave;                // 384*37 tasks exactly
    const int bh = task / QTILES;
    const int qt = task - bh * QTILES;
    const int t0 = qt * 16;
    const int g8 = (lane >> 4) << 3;

    // Q A-fragments, d-chunks 0..31 / 32..63 (loaded once)
    const long qbase = ((long)bh * NPAD + t0 + (lane & 15)) * HD + g8;
    bf16x16 qa0 = load_frag16(Qb + qbase);
    bf16x16 qa1 = load_frag16(Qb + qbase + 32);

    f32x8 oacc[4] = {};
    float mrow[8], lrow[8];
#pragma unroll
    for (int r = 0; r < 8; ++r) { mrow[r] = -1e30f; lrow[r] = 0.f; }

    int p = 0;
    for (int k0 = 0; k0 < NPAD; k0 += 32, p ^= 1) {
        // --- async global->LDS staging of the 32x64 K tile (ASYNCcnt path) ---
        {
            const __bf16* ksrc = Kb + ((long)bh * NPAD + k0 + lane) * HD;
            const unsigned klds = (unsigned)(size_t)&sK[wave][p][lane * HD];
#pragma unroll
            for (int j = 0; j < 8; ++j)
                asm volatile("global_load_async_to_lds_b128 %0, %1, off"
                             :: "v"(klds + j * 16), "v"(ksrc + j * 8) : "memory");
            asm volatile("s_wait_asynccnt 0x0" ::: "memory");
        }

        // S = Q . K^T : two 16x16 sub-tiles, K-loop over d (2 x 32)
        f32x8 S[2] = {};
#pragma unroll
        for (int sub = 0; sub < 2; ++sub) {
            const __bf16* kt = &sK[wave][p][(sub * 16 + (lane & 15)) * HD + g8];
            S[sub] = wmma_bf16(qa0, load_frag16(kt),      S[sub]);
            S[sub] = wmma_bf16(qa1, load_frag16(kt + 32), S[sub]);
        }
        // mask padded key tokens
#pragma unroll
        for (int sub = 0; sub < 2; ++sub) {
            const int tok = k0 + sub * 16 + (lane & 15);
            if (tok >= NTOK) {
#pragma unroll
                for (int r = 0; r < 8; ++r) S[sub][r] = -1e30f;
            }
        }

        // online softmax per row; P written to LDS for C->A fragment re-layout
#pragma unroll
        for (int r = 0; r < 8; ++r) {
            float lm = fmaxf(S[0][r], S[1][r]);
#pragma unroll
            for (int msk = 1; msk < 16; msk <<= 1) lm = fmaxf(lm, __shfl_xor(lm, msk, 32));
            const float nm = fmaxf(mrow[r], lm);
            const float alpha = __expf(mrow[r] - nm);
            const float p0 = __expf(S[0][r] - nm);
            const float p1 = __expf(S[1][r] - nm);
            float ls = p0 + p1;
#pragma unroll
            for (int msk = 1; msk < 16; msk <<= 1) ls += __shfl_xor(ls, msk, 32);
            lrow[r] = lrow[r] * alpha + ls;
            mrow[r] = nm;
#pragma unroll
            for (int nt = 0; nt < 4; ++nt) oacc[nt][r] *= alpha;
            const int prow = r + g8;
            sP[wave][prow * 32 + (lane & 15)]      = (__bf16)p0;
            sP[wave][prow * 32 + 16 + (lane & 15)] = (__bf16)p1;
        }

        // P A-fragment from LDS (rows x K=32)
        bf16x16 pa = load_frag16(&sP[wave][(lane & 15) * 32 + g8]);

        // O += P . V : V stored transposed -> B-fragments are contiguous b128 loads
#pragma unroll
        for (int nt = 0; nt < 4; ++nt) {
            const __bf16* vt = VbT + ((long)bh * HD + nt * 16 + (lane & 15)) * NPAD + k0 + g8;
            oacc[nt] = wmma_bf16(pa, load_frag16(vt), oacc[nt]);
        }
    }

    // epilogue: normalize and scatter to [B*N, CH] bf16 for the proj GEMM
    const int b = bh / NUM_H;
    const int h = bh - b * NUM_H;
#pragma unroll
    for (int r = 0; r < 8; ++r) {
        const int t = t0 + r + g8;
        if (t < NTOK) {
            const float inv = 1.f / lrow[r];
            const long obase = ((long)(b * NTOK + t)) * CH + h * HD + (lane & 15);
#pragma unroll
            for (int nt = 0; nt < 4; ++nt)
                Ob[obase + nt * 16] = (__bf16)(oacc[nt][r] * inv);
        }
    }
}

// ---------------- output projection ----------------
__global__ void proj_kernel(const __bf16* __restrict__ Ob,
                            const __bf16* __restrict__ wB,
                            const float* __restrict__ proj_b,
                            float* __restrict__ out) {
    const int lane = threadIdx.x & 31;
    const int wave = threadIdx.x >> 5;
    const int m0 = blockIdx.x * 32;
    const int o0 = (blockIdx.y * 8 + wave) * 32;

    f32x8 acc[4] = {};
    gemm_tile2x2(Ob, wB, m0, o0, CH, lane, acc);

    const int rowoff = (lane >> 4) << 3;
#pragma unroll
    for (int ot = 0; ot < 2; ++ot) {
        const int col = o0 + ot * 16 + (lane & 15);
        const float bias = proj_b[col];
#pragma unroll
        for (int mt = 0; mt < 2; ++mt) {
#pragma unroll
            for (int r = 0; r < 8; ++r) {
                const int m = m0 + mt * 16 + r + rowoff;
                out[(long)m * CH + col] = acc[mt * 2 + ot][r] + bias;
            }
        }
    }
}

// ---------------- host launch ----------------
extern "C" void kernel_launch(void* const* d_in, const int* in_sizes, int n_in,
                              void* d_out, int out_size, void* d_ws, size_t ws_size,
                              hipStream_t stream) {
    (void)in_sizes; (void)n_in; (void)out_size; (void)ws_size;
    const float* x      = (const float*)d_in[0];
    const float* rope   = (const float*)d_in[1];
    const float* qkv_w  = (const float*)d_in[2];
    const float* q_bias = (const float*)d_in[3];
    const float* v_bias = (const float*)d_in[4];
    const float* proj_w = (const float*)d_in[5];
    const float* proj_b = (const float*)d_in[6];
    float* out = (float*)d_out;

    // workspace carve-up (all sizes multiples of 256 B)
    char* ws = (char*)d_ws;
    const long nX   = (long)M_ROWS * CH;               // 14,180,352
    const long nQW  = (long)3 * CH * CH;               //  1,769,472
    const long nPW  = (long)CH * CH;                   //    589,824
    const long nQKV = (long)BATCH * NUM_H * NPAD * HD; // 14,942,208 per buffer

    __bf16* xB  = (__bf16*)ws;              ws += nX * 2;
    __bf16* qwB = (__bf16*)ws;              ws += nQW * 2;
    __bf16* pwB = (__bf16*)ws;              ws += nPW * 2;
    __bf16* Qb  = (__bf16*)ws;              ws += nQKV * 2;
    __bf16* Kb  = (__bf16*)ws;              ws += nQKV * 2;
    __bf16* VbT = (__bf16*)ws;              ws += nQKV * 2;
    __bf16* Ob  = (__bf16*)ws;              ws += nX * 2;

    cvt_f32_bf16<<<2048, 256, 0, stream>>>(x,      xB,  nX);
    cvt_f32_bf16<<<1024, 256, 0, stream>>>(qkv_w,  qwB, nQW);
    cvt_f32_bf16<<<512,  256, 0, stream>>>(proj_w, pwB, nPW);
    zero_bf16   <<<2048, 256, 0, stream>>>(Qb, 3 * nQKV);  // Q,K,V^T contiguous

    qkv_rope_kernel<<<dim3(M_ROWS / 32, (3 * CH) / 256), 256, 0, stream>>>(
        xB, qwB, q_bias, v_bias, rope, Qb, Kb, VbT);

    attn_kernel<<<dim3((BATCH * NUM_H * QTILES) / 8), 256, 0, stream>>>(Qb, Kb, VbT, Ob);

    proj_kernel<<<dim3(M_ROWS / 32, CH / 256), 256, 0, stream>>>(Ob, pwB, proj_b, out);
}
